// CrossAttention_49151605735826
// MI455X (gfx1250) — matrix operand order repaired and checked by reading, not compiled, not run on hardware
//
#include <hip/hip_runtime.h>

// ---------------------------------------------------------------------------
// CDNA5 (gfx1250, wave32) cross-attention forward.
// B=2, D=1024, S_KV=S_Q=2048, H=16, hd=64.
// Matmuls: v_wmma_f32_16x16x32_bf16 (f32 accum).
// Attention K/V tiles staged by the Tensor Data Mover (tensor_load_to_lds),
// double-buffered, synchronized with s_wait_tensorcnt.
// ---------------------------------------------------------------------------

typedef __attribute__((ext_vector_type(16))) __bf16 v16bf;
typedef __attribute__((ext_vector_type(8)))  float  v8f;
typedef __attribute__((ext_vector_type(4)))  unsigned int u32x4;
typedef __attribute__((ext_vector_type(8)))  int i32x8;
typedef __attribute__((ext_vector_type(4)))  int i32x4;

#define CB  2
#define CD  1024
#define CS  2048
#define CH  16
#define CHD 64

#define BM 128
#define BN 128
#define BK 32
#define LDT 40   // LDS row stride (bf16 elems) for k=32 tiles: 80B, 16B-aligned
#define LDA 72   // LDS row stride for 64-wide attention tiles: 144B, 16B-aligned

static __device__ __forceinline__ v8f zero8() {
  v8f z = {0.f, 0.f, 0.f, 0.f, 0.f, 0.f, 0.f, 0.f};
  return z;
}

static __device__ __forceinline__ v8f wmma_bf16(v16bf a, v16bf b, v8f c) {
  // (neg_a, A, neg_b, B, c_mod, C, reuse_a, reuse_b)
  return __builtin_amdgcn_wmma_f32_16x16x32_bf16(false, a, false, b,
                                                 (short)0, c, false, false);
}

// A-fragment (16x32 bf16): lane L holds row M=L%16, K in {kb..kb+7, 16+kb..16+kb+7},
// kb = 8*(L/16). Source tile is row-major [m][k] with row stride `ld`.
static __device__ __forceinline__ v16bf load_frag_a(const __bf16* tile, int ld,
                                                    int m0, int kofs) {
  const int lane = threadIdx.x & 31;
  const __bf16* r = tile + (size_t)(m0 + (lane & 15)) * ld + kofs + ((lane >> 4) << 3);
  union { uint4 u[2]; v16bf v; } x;
  x.u[0] = *reinterpret_cast<const uint4*>(r);        // K = kb .. kb+7
  x.u[1] = *reinterpret_cast<const uint4*>(r + 16);   // K = 16+kb .. 16+kb+7
  return x.v;
}

// B-fragment (32x16 bf16, KxN): lane L holds col N=L%16, K in {kb..kb+7, 16+kb..16+kb+7}, kb = 8*(L/16).
// Source tile is [n][k] layout, stride `ld`.
static __device__ __forceinline__ v16bf load_frag_b(const __bf16* tile, int ld,
                                                    int n0, int kofs) {
  const int lane = threadIdx.x & 31;
  const __bf16* r = tile + (size_t)(n0 + (lane & 15)) * ld + kofs + ((lane >> 4) << 3);
  union { uint4 u[2]; v16bf v; } x;
  x.u[0] = *reinterpret_cast<const uint4*>(r);
  x.u[1] = *reinterpret_cast<const uint4*>(r + 16);
  return x.v;
}

// ---------------------------------------------------------------------------
// TDM: 2D tile (bf16) from global -> LDS. Descriptor per CDNA5 ISA ch. 8.
// Pad feature: insert 16B after every 128B row so LDS row stride = 144B (LDA=72).
// tile_d0 must be 64 (one 128B row) for this pad configuration.
// This toolchain's builtin takes 6 args (g0, g1, g2, g3, extra group, cpol).
// ---------------------------------------------------------------------------
static __device__ __forceinline__ void tdm_load_2d_bf16(
    unsigned lds_addr, const void* gaddr, unsigned tensor_d0, unsigned tensor_d1,
    unsigned tile_d0, unsigned tile_d1, unsigned long long stride0) {
  const unsigned long long ga = (unsigned long long)(uintptr_t)gaddr;
  u32x4 g0;
  g0[0] = 1u;                                   // count=1 (valid user descriptor)
  g0[1] = lds_addr;                             // LDS byte address
  g0[2] = (unsigned)ga;                         // global addr [95:64]
  g0[3] = (unsigned)((ga >> 32) & 0x1FFFFFFu)   // global addr [120:96]
          | 0x80000000u;                        // type=2 ("image") in [127:126]
  i32x8 g1;
  g1[0] = (int)((1u << 16)                      // data_size = 1 -> 2 bytes
                | (1u << 20)                    // pad_enable
                | (4u << 22)                    // pad_interval: 2^(4+1)=32 dwords=128B
                | (3u << 25));                  // pad_amount: 3 -> 4 dwords = 16B
  g1[1] = (int)((tensor_d0 & 0xFFFFu) << 16);   // tensor_dim0 [79:48] low half
  g1[2] = (int)(((tensor_d0 >> 16) & 0xFFFFu) |
                ((tensor_d1 & 0xFFFFu) << 16)); // tensor_dim0 hi / tensor_dim1 lo
  g1[3] = (int)(((tensor_d1 >> 16) & 0xFFFFu) |
                ((tile_d0 & 0xFFFFu) << 16));   // tensor_dim1 hi / tile_dim0
  g1[4] = (int)(tile_d1 & 0xFFFFu);             // tile_dim1; tile_dim2 = 0 (2D)
  g1[5] = (int)(unsigned)stride0;               // tensor_dim0_stride lo32
  g1[6] = (int)((unsigned)(stride0 >> 32) & 0xFFFFu);  // stride hi16; dim1_stride=0
  g1[7] = 0;
  const i32x4 z4 = {0, 0, 0, 0};                // groups 2/3 unused for 2D tensors
  const i32x8 z8 = {0, 0, 0, 0, 0, 0, 0, 0};
  __builtin_amdgcn_tensor_load_to_lds(g0, g1, z4, z4, z8, 0);
}

static __device__ __forceinline__ unsigned lds_addr_of(const void* p) {
  return (unsigned)(uintptr_t)p;  // generic LDS pointer: low 32 bits = LDS byte addr
}

// ---------------------------------------------------------------------------
// GEMM 1/2: C[m,n] = sum_k X[b,k,m] * W[k,n]   (X is [D,S] f32, W is [K,N] f32)
// EPI==0: scatter n<1024 -> K [B,H,S,64] bf16, n>=1024 -> V^T [B,H,64,S] bf16
// EPI==1: Q [B,H,S,64] bf16, scaled by 0.125 (= hd^-0.5)
// ---------------------------------------------------------------------------
template <int EPI>
__global__ __launch_bounds__(256) void gemm_xt_w(const float* __restrict__ X,
                                                 const float* __restrict__ W,
                                                 __bf16* __restrict__ out0,
                                                 __bf16* __restrict__ out1,
                                                 int S, int N, int K) {
  __shared__ __attribute__((aligned(16))) __bf16 As[BM * LDT];
  __shared__ __attribute__((aligned(16))) __bf16 Bs[BN * LDT];

  const int bz = blockIdx.z;
  const int m0 = blockIdx.y * BM;
  const int n0 = blockIdx.x * BN;
  const int t    = threadIdx.x;
  const int lane = t & 31;
  const int wave = t >> 5;
  const int wm = (wave & 3) * 32;   // 4 waves along M
  const int wn = (wave >> 2) * 64;  // 2 waves along N

  const float* Xb = X + (size_t)bz * K * S;

  v8f acc[2][4];
#pragma unroll
  for (int mt = 0; mt < 2; ++mt)
#pragma unroll
    for (int nt = 0; nt < 4; ++nt) acc[mt][nt] = zero8();

  const int mloc = (t & 31) << 2;  // 0..124, float4 along the contiguous dim
  const int krow = t >> 5;         // 0..7

  for (int kt = 0; kt < K / BK; ++kt) {
    // prefetch next k-tile while this one is staged/consumed
    if (kt + 1 < K / BK) {
      __builtin_prefetch(Xb + (size_t)((kt + 1) * BK + krow) * S + m0 + mloc, 0, 1);
      __builtin_prefetch(W + (size_t)((kt + 1) * BK + krow) * N + n0 + mloc, 0, 1);
    }
    // ---- stage A: X[k][m] -> As[m][k] (transpose + f32->bf16) ----
#pragma unroll
    for (int p = 0; p < 4; ++p) {
      const int k = krow + p * 8;
      float4 f = *reinterpret_cast<const float4*>(
          Xb + (size_t)(kt * BK + k) * S + m0 + mloc);
      As[(mloc + 0) * LDT + k] = (__bf16)f.x;
      As[(mloc + 1) * LDT + k] = (__bf16)f.y;
      As[(mloc + 2) * LDT + k] = (__bf16)f.z;
      As[(mloc + 3) * LDT + k] = (__bf16)f.w;
    }
    // ---- stage B: W[k][n] -> Bs[n][k] ----
#pragma unroll
    for (int p = 0; p < 4; ++p) {
      const int k = krow + p * 8;
      float4 f = *reinterpret_cast<const float4*>(
          W + (size_t)(kt * BK + k) * N + n0 + mloc);
      Bs[(mloc + 0) * LDT + k] = (__bf16)f.x;
      Bs[(mloc + 1) * LDT + k] = (__bf16)f.y;
      Bs[(mloc + 2) * LDT + k] = (__bf16)f.z;
      Bs[(mloc + 3) * LDT + k] = (__bf16)f.w;
    }
    __syncthreads();

    v16bf a0 = load_frag_a(As, LDT, wm + 0, 0);
    v16bf a1 = load_frag_a(As, LDT, wm + 16, 0);
#pragma unroll
    for (int nt = 0; nt < 4; ++nt) {
      v16bf bb = load_frag_b(Bs, LDT, wn + nt * 16, 0);
      acc[0][nt] = wmma_bf16(a0, bb, acc[0][nt]);
      acc[1][nt] = wmma_bf16(a1, bb, acc[1][nt]);
    }
    __syncthreads();
  }

  // ---- epilogue ----
#pragma unroll
  for (int mt = 0; mt < 2; ++mt)
#pragma unroll
    for (int nt = 0; nt < 4; ++nt)
#pragma unroll
      for (int r = 0; r < 8; ++r) {
        const int M  = m0 + wm + mt * 16 + r + ((lane >> 4) << 3);
        const int Nn = n0 + wn + nt * 16 + (lane & 15);
        const float val = acc[mt][nt][r];
        if (EPI == 0) {
          if (Nn < CD) {  // K part: [B,H,S,64]
            const int h = Nn >> 6, d2 = Nn & 63;
            out0[(((size_t)bz * CH + h) * S + M) * CHD + d2] = (__bf16)val;
          } else {        // V part, stored transposed: [B,H,64,S]
            const int n2 = Nn - CD;
            const int h = n2 >> 6, d2 = n2 & 63;
            out1[(((size_t)bz * CH + h) * CHD + d2) * S + M] = (__bf16)val;
          }
        } else {          // Q, pre-scaled by hd^-0.5
          const int h = Nn >> 6, d2 = Nn & 63;
          out0[(((size_t)bz * CH + h) * S + M) * CHD + d2] =
              (__bf16)(val * 0.125f);
        }
      }
}

// ---------------------------------------------------------------------------
// Flash attention: per (b,h), q-tile of 128 rows per block, 16 rows per wave.
// K/V tiles DMA'd into double-buffered LDS by the Tensor Data Mover.
// ---------------------------------------------------------------------------
__global__ __launch_bounds__(256) void attn_kernel(const __bf16* __restrict__ Q,
                                                   const __bf16* __restrict__ Kw,
                                                   const __bf16* __restrict__ VT,
                                                   __bf16* __restrict__ AO,
                                                   int SQ, int SKV) {
  __shared__ __attribute__((aligned(16))) __bf16 Kt0[64 * LDA];      // [kv][hd]
  __shared__ __attribute__((aligned(16))) __bf16 Kt1[64 * LDA];
  __shared__ __attribute__((aligned(16))) __bf16 Vt0[64 * LDA];      // [hd][kv]
  __shared__ __attribute__((aligned(16))) __bf16 Vt1[64 * LDA];
  __shared__ __attribute__((aligned(16))) __bf16 Pt[8 * 16 * LDA];   // per-wave P

  const int t    = threadIdx.x;
  const int lane = t & 31;
  const int wave = t >> 5;
  const int qt = blockIdx.x, h = blockIdx.y, bz = blockIdx.z;
  const int bh = bz * CH + h;
  const int q0 = qt * 128 + wave * 16;

  // Q fragments straight from global (layout [bh][s][64] == row-major [m][k])
  const __bf16* Qbase = Q + ((size_t)bh * SQ + q0) * CHD;
  const v16bf qf0 = load_frag_a(Qbase, CHD, 0, 0);
  const v16bf qf1 = load_frag_a(Qbase, CHD, 0, 32);

  v8f o[4];
#pragma unroll
  for (int nt = 0; nt < 4; ++nt) o[nt] = zero8();
  v8f m_run, l_run;
#pragma unroll
  for (int r = 0; r < 8; ++r) { m_run[r] = -1e30f; l_run[r] = 0.f; }

  __bf16* Pw = &Pt[wave * 16 * LDA];

  const __bf16* Kbase = Kw + (size_t)bh * SKV * CHD;   // [SKV][64]
  const __bf16* Vbase = VT + (size_t)bh * CHD * SKV;   // [64][SKV]
  const unsigned ldsK[2] = {lds_addr_of(Kt0), lds_addr_of(Kt1)};
  const unsigned ldsV[2] = {lds_addr_of(Vt0), lds_addr_of(Vt1)};

  const int NT = SKV / 64;
  // prologue: DMA tile 0 into buffer 0
  if (wave == 0) {
    tdm_load_2d_bf16(ldsK[0], Kbase, CHD, 64, CHD, 64, CHD);
    tdm_load_2d_bf16(ldsV[0], Vbase, (unsigned)SKV, CHD, 64, CHD,
                     (unsigned long long)SKV);
  }

  for (int it = 0; it < NT; ++it) {
    if (wave == 0) __builtin_amdgcn_s_wait_tensorcnt(0);
    __syncthreads();  // buffer it&1 now holds tile `it` for all waves

    // overlap: DMA tile it+1 into the other buffer while we compute
    if (wave == 0 && it + 1 < NT) {
      const int kv1 = (it + 1) * 64;
      tdm_load_2d_bf16(ldsK[(it + 1) & 1], Kbase + (size_t)kv1 * CHD,
                       CHD, 64, CHD, 64, CHD);
      tdm_load_2d_bf16(ldsV[(it + 1) & 1], Vbase + kv1,
                       (unsigned)(SKV - kv1), CHD, 64, CHD,
                       (unsigned long long)SKV);
    }

    const __bf16* Ktile = (it & 1) ? Kt1 : Kt0;
    const __bf16* Vtile = (it & 1) ? Vt1 : Vt0;

    // S = Q @ K^T  (B-fragment [n=kv][k=hd] is exactly the Ktile layout)
    v8f s[4];
#pragma unroll
    for (int nt = 0; nt < 4; ++nt) s[nt] = zero8();
#pragma unroll
    for (int ks = 0; ks < 2; ++ks) {
      const v16bf a = ks ? qf1 : qf0;
#pragma unroll
      for (int nt = 0; nt < 4; ++nt) {
        v16bf bb = load_frag_b(Ktile, LDA, nt * 16, ks * 32);
        s[nt] = wmma_bf16(a, bb, s[nt]);
      }
    }

    // online softmax: rows live across 16-lane halves (wave32)
    v8f mx = s[0];
#pragma unroll
    for (int nt = 1; nt < 4; ++nt)
#pragma unroll
      for (int r = 0; r < 8; ++r) mx[r] = fmaxf(mx[r], s[nt][r]);
#pragma unroll
    for (int d = 1; d <= 8; d <<= 1)
#pragma unroll
      for (int r = 0; r < 8; ++r)
        mx[r] = fmaxf(mx[r], __shfl_xor(mx[r], d, 32));

    v8f mnew, alpha, rs;
#pragma unroll
    for (int r = 0; r < 8; ++r) {
      mnew[r]  = fmaxf(m_run[r], mx[r]);
      alpha[r] = __expf(m_run[r] - mnew[r]);
      rs[r]    = 0.f;
    }
#pragma unroll
    for (int nt = 0; nt < 4; ++nt)
#pragma unroll
      for (int r = 0; r < 8; ++r) {
        const float p = __expf(s[nt][r] - mnew[r]);
        s[nt][r] = p;
        rs[r] += p;
      }
#pragma unroll
    for (int d = 1; d <= 8; d <<= 1)
#pragma unroll
      for (int r = 0; r < 8; ++r) rs[r] += __shfl_xor(rs[r], d, 32);
#pragma unroll
    for (int r = 0; r < 8; ++r) {
      l_run[r] = l_run[r] * alpha[r] + rs[r];
      m_run[r] = mnew[r];
    }
#pragma unroll
    for (int nt = 0; nt < 4; ++nt)
#pragma unroll
      for (int r = 0; r < 8; ++r) o[nt][r] *= alpha[r];

    // P: C-layout -> A-layout via per-wave LDS round trip
#pragma unroll
    for (int nt = 0; nt < 4; ++nt)
#pragma unroll
      for (int r = 0; r < 8; ++r)
        Pw[(r + ((lane >> 4) << 3)) * LDA + nt * 16 + (lane & 15)] =
            (__bf16)s[nt][r];

    // O += P @ V  (B-fragment [n=hd][k=kv] is exactly the Vtile layout)
#pragma unroll
    for (int ks = 0; ks < 2; ++ks) {
      v16bf a = load_frag_a(Pw, LDA, 0, ks * 32);
#pragma unroll
      for (int nt = 0; nt < 4; ++nt) {
        v16bf bb = load_frag_b(Vtile, LDA, nt * 16, ks * 32);
        o[nt] = wmma_bf16(a, bb, o[nt]);
      }
    }
  }

  // normalize and store attention output [B, S_Q, D] bf16
  __bf16* ao = AO + ((size_t)bz * SQ + q0) * CD + h * CHD;
  v8f inv;
#pragma unroll
  for (int r = 0; r < 8; ++r) inv[r] = 1.0f / l_run[r];
#pragma unroll
  for (int nt = 0; nt < 4; ++nt)
#pragma unroll
    for (int r = 0; r < 8; ++r) {
      const int M = r + ((lane >> 4) << 3);
      ao[(size_t)M * CD + nt * 16 + (lane & 15)] = (__bf16)(o[nt][r] * inv[r]);
    }
}

// ---------------------------------------------------------------------------
// GEMM 3: out[b, n, m] = sum_k AO[b, m, k] * W3[k, n]   (AO bf16, out f32)
// ---------------------------------------------------------------------------
__global__ __launch_bounds__(256) void gemm_ao_w3(const __bf16* __restrict__ AO,
                                                  const float* __restrict__ W3,
                                                  float* __restrict__ out,
                                                  int S, int N, int K) {
  __shared__ __attribute__((aligned(16))) __bf16 As[BM * LDT];
  __shared__ __attribute__((aligned(16))) __bf16 Bs[BN * LDT];

  const int bz = blockIdx.z;
  const int m0 = blockIdx.y * BM;
  const int n0 = blockIdx.x * BN;
  const int t    = threadIdx.x;
  const int lane = t & 31;
  const int wave = t >> 5;
  const int wm = (wave & 3) * 32;
  const int wn = (wave >> 2) * 64;

  v8f acc[2][4];
#pragma unroll
  for (int mt = 0; mt < 2; ++mt)
#pragma unroll
    for (int nt = 0; nt < 4; ++nt) acc[mt][nt] = zero8();

  const int arow = t >> 1;          // 0..127
  const int akh  = (t & 1) << 4;    // 0 or 16
  const int nloc = (t & 31) << 2;
  const int krow = t >> 5;

  for (int kt = 0; kt < K / BK; ++kt) {
    if (kt + 1 < K / BK) {
      __builtin_prefetch(AO + ((size_t)bz * S + m0 + arow) * K + (kt + 1) * BK + akh,
                         0, 1);
      __builtin_prefetch(W3 + (size_t)((kt + 1) * BK + krow) * N + n0 + nloc, 0, 1);
    }
    // A already bf16 row-major: straight 32B copies
    {
      const __bf16* ag =
          AO + ((size_t)bz * S + m0 + arow) * K + kt * BK + akh;
      uint4 u0 = *reinterpret_cast<const uint4*>(ag);
      uint4 u1 = *reinterpret_cast<const uint4*>(ag + 8);
      *reinterpret_cast<uint4*>(&As[arow * LDT + akh])     = u0;
      *reinterpret_cast<uint4*>(&As[arow * LDT + akh + 8]) = u1;
    }
    // B: W3[k][n] -> Bs[n][k]
#pragma unroll
    for (int p = 0; p < 4; ++p) {
      const int k = krow + p * 8;
      float4 f = *reinterpret_cast<const float4*>(
          W3 + (size_t)(kt * BK + k) * N + n0 + nloc);
      Bs[(nloc + 0) * LDT + k] = (__bf16)f.x;
      Bs[(nloc + 1) * LDT + k] = (__bf16)f.y;
      Bs[(nloc + 2) * LDT + k] = (__bf16)f.z;
      Bs[(nloc + 3) * LDT + k] = (__bf16)f.w;
    }
    __syncthreads();

    v16bf a0 = load_frag_a(As, LDT, wm + 0, 0);
    v16bf a1 = load_frag_a(As, LDT, wm + 16, 0);
#pragma unroll
    for (int nt = 0; nt < 4; ++nt) {
      v16bf bb = load_frag_b(Bs, LDT, wn + nt * 16, 0);
      acc[0][nt] = wmma_bf16(a0, bb, acc[0][nt]);
      acc[1][nt] = wmma_bf16(a1, bb, acc[1][nt]);
    }
    __syncthreads();
  }

  // transposed f32 store: out[b, n, m]
#pragma unroll
  for (int mt = 0; mt < 2; ++mt)
#pragma unroll
    for (int nt = 0; nt < 4; ++nt)
#pragma unroll
      for (int r = 0; r < 8; ++r) {
        const int M  = m0 + wm + mt * 16 + r + ((lane >> 4) << 3);
        const int Nn = n0 + wn + nt * 16 + (lane & 15);
        out[((size_t)bz * N + Nn) * S + M] = acc[mt][nt][r];
      }
}

// ---------------------------------------------------------------------------
extern "C" void kernel_launch(void* const* d_in, const int* in_sizes, int n_in,
                              void* d_out, int out_size, void* d_ws,
                              size_t ws_size, hipStream_t stream) {
  const float* code   = (const float*)d_in[0];  // [B, D, S_KV]
  const float* inputs = (const float*)d_in[1];  // [B, D, S_Q]
  const float* W1     = (const float*)d_in[2];  // [D, 2D]
  const float* W2     = (const float*)d_in[3];  // [D, D]
  const float* W3     = (const float*)d_in[4];  // [D, D]
  float* out = (float*)d_out;                   // [B, D, S_Q]

  const size_t seg = (size_t)CB * CH * CS * CHD * sizeof(__bf16);  // 8 MiB
  char* ws = (char*)d_ws;
  __bf16* Kws  = (__bf16*)(ws);            // [B,H,S_KV,64]
  __bf16* VTws = (__bf16*)(ws + seg);      // [B,H,64,S_KV]
  __bf16* Qws  = (__bf16*)(ws + 2 * seg);  // [B,H,S_Q,64]
  __bf16* AOws = (__bf16*)(ws + 3 * seg);  // [B,S_Q,D]

  dim3 blk(256);
  // KV projection: M=S_KV=2048, N=2D=2048, K=D=1024
  gemm_xt_w<0><<<dim3(CD * 2 / BN, CS / BM, CB), blk, 0, stream>>>(
      code, W1, Kws, VTws, CS, 2 * CD, CD);
  // Q projection: M=S_Q=2048, N=D=1024, K=D=1024
  gemm_xt_w<1><<<dim3(CD / BN, CS / BM, CB), blk, 0, stream>>>(
      inputs, W2, Qws, nullptr, CS, CD, CD);
  // attention: 128 q-rows per block, per (b,h)
  attn_kernel<<<dim3(CS / 128, CH, CB), blk, 0, stream>>>(Qws, Kws, VTws, AOws,
                                                          CS, CS);
  // output projection + transpose
  gemm_ao_w3<<<dim3(CD / BN, CS / BM, CB), blk, 0, stream>>>(AOws, W3, out, CS,
                                                             CD, CD);
}
